// Attention_1786706395201
// MI455X (gfx1250) — matrix operand order repaired and checked by reading, not compiled, not run on hardware
//
#include <hip/hip_runtime.h>
#include <hip/hip_bf16.h>
#include <math.h>

// ---------------------------------------------------------------------------
// Types / WMMA fragment helpers (CDNA5 / gfx1250, wave32)
// ---------------------------------------------------------------------------
typedef __attribute__((ext_vector_type(16))) __bf16       v16bf;
typedef __attribute__((ext_vector_type(8)))  float        v8f;
typedef __attribute__((ext_vector_type(4)))  unsigned int u32x4;
typedef __attribute__((ext_vector_type(8)))  int          i32x8;
typedef __attribute__((ext_vector_type(4)))  int          i32x4;

__device__ __forceinline__ v8f wmma_bf16(v16bf a, v16bf b, v8f c) {
  // (neg_a, A, neg_b, B, c_mod, C, reuse_a, reuse_b)
  return __builtin_amdgcn_wmma_f32_16x16x32_bf16(false, a, false, b,
                                                 (short)0, c, false, false);
}

// A tile 16x32 (MxK) bf16 row-major, leading dim ld.
// lane<16 -> M=lane, K slots {0..7,16..23}; lane>=16 -> M=lane-16, {8..15,24..31}
__device__ __forceinline__ v16bf load_a_bf(const __bf16* A, int ld) {
  const int l = threadIdx.x & 31;
  const __bf16* p0 = A + (l & 15) * ld + (l >> 4) * 8;
  v16bf a;
#pragma unroll
  for (int i = 0; i < 8; ++i) { a[i] = p0[i]; a[i + 8] = p0[16 + i]; }
  return a;
}

// Same but f32 source, convert in registers.
__device__ __forceinline__ v16bf load_a_f32(const float* A, int ld) {
  const int l = threadIdx.x & 31;
  const float* p0 = A + (l & 15) * ld + (l >> 4) * 8;
  v16bf a;
#pragma unroll
  for (int i = 0; i < 8; ++i) {
    a[i]     = (__bf16)p0[i];
    a[i + 8] = (__bf16)p0[16 + i];
  }
  return a;
}

// B tile 32x16 (KxN) from row-major weight Wt[N][K] (GEMM is C = A x Wt^T).
// lanes 0-15: K {0..7,16..23}, lanes 16-31: K {8..15,24..31}, column = lane&15.
__device__ __forceinline__ v16bf load_b_bf(const __bf16* Wt, int ld) {
  const int l = threadIdx.x & 31;
  const __bf16* p = Wt + (l & 15) * ld + (l >> 4) * 8;
  v16bf b;
#pragma unroll
  for (int i = 0; i < 8; ++i) { b[i] = p[i]; b[i + 8] = p[16 + i]; }
  return b;
}

// C/D 16x16 f32: VGPR i -> row (i + 8*(lane>=16)), col = lane&15.
__device__ __forceinline__ void store_c(float* C, int ld, v8f c) {
  const int l  = threadIdx.x & 31;
  const int mo = (l >> 4) * 8;
  const int n  = l & 15;
#pragma unroll
  for (int i = 0; i < 8; ++i) C[(mo + i) * ld + n] = c[i];
}

// Rows 0..7 only (batch 8 padded to 16): lanes 0-15 hold them.
__device__ __forceinline__ void store_c_lo(float* C, int ld, v8f c) {
  const int l = threadIdx.x & 31;
  if (l < 16) {
#pragma unroll
    for (int i = 0; i < 8; ++i) C[i * ld + l] = c[i];
  }
}

// ---------------------------------------------------------------------------
// Hardware tanh (CDNA5 TRANS op). s_delay_alu covers the TRANS RAW hazard.
// ---------------------------------------------------------------------------
__device__ __forceinline__ float tanh_hw(float x) {
  float r;
  asm volatile("v_tanh_f32 %0, %1\n\t"
               "s_delay_alu instid0(TRANS32_DEP_1)"
               : "=v"(r) : "v"(x));
  return r;
}

__device__ __forceinline__ float sigmoid_f(float x) {
  return 1.f / (1.f + __expf(-x));
}

// ---------------------------------------------------------------------------
// Tensor Data Mover: 1-D contiguous global -> LDS copy (data_size = 8B units).
// Descriptor per CDNA5 ISA ch.8 (group0: count/lds/global/type, group1: dims).
// ---------------------------------------------------------------------------
__device__ __forceinline__ unsigned lds_addr_of(const void* p) {
  return (unsigned)(unsigned long long)
      (__attribute__((address_space(3))) const void*)p;
}

__device__ __forceinline__ void tdm_stage_1d(const void* gsrc,
                                             unsigned lds_byte_addr,
                                             unsigned nbytes) {
  const unsigned long long ga = (unsigned long long)gsrc;
  const unsigned units = nbytes >> 3;                        // 8-byte units
  u32x4 g0;
  g0[0] = 1u;                                                // count=1
  g0[1] = lds_byte_addr;                                     // lds_addr
  g0[2] = (unsigned)ga;                                      // global_addr lo
  g0[3] = (unsigned)((ga >> 32) & 0x1FFFFFFull) | (2u << 30);// addr hi | type=2
  i32x8 g1;
  g1[0] = (3 << 16);                                         // data_size=3 (8B)
  g1[1] = (int)((units & 0xFFFFu) << 16);                    // tensor_dim0 lo16
  g1[2] = (int)((units >> 16) & 0xFFFFu) | (1 << 16);        // dim0 hi | dim1=1
  g1[3] = (int)(units << 16);                                // tile_dim0
  g1[4] = 0;                                                 // tile_dim1/2 = 0
  g1[5] = (int)units;                                        // dim0_stride lo
  g1[6] = 0;
  g1[7] = 0;
  i32x4 z4 = {0, 0, 0, 0};
#if __clang_major__ >= 23
  i32x8 z8 = {0, 0, 0, 0, 0, 0, 0, 0};
  __builtin_amdgcn_tensor_load_to_lds(g0, g1, z4, z4, z8, 0);
#else
  __builtin_amdgcn_tensor_load_to_lds(g0, g1, z4, z4, 0);
#endif
}

// ---------------------------------------------------------------------------
// f32 -> bf16 conversion
// ---------------------------------------------------------------------------
__global__ void cvt_kernel(const float* __restrict__ src,
                           __bf16* __restrict__ dst, int count) {
  int i = blockIdx.x * blockDim.x + threadIdx.x;
  if (i < count) dst[i] = (__bf16)src[i];
}

// ---------------------------------------------------------------------------
// gi = inputs @ Wih^T for ALL timesteps (hoisted out of the recurrence).
// A [2400][128] f32, Wih [768][128] bf16, gi [2400][768] f32.
// One 16-row tile per block, 8 waves x 6 col-tiles, K=128 -> 4 WMMA each.
// ---------------------------------------------------------------------------
__global__ __launch_bounds__(256)
void gi_kernel(const float* __restrict__ A,
               const __bf16* __restrict__ Wih,
               float* __restrict__ gi) {
  const int mt   = blockIdx.x;            // 0..149
  const int wave = threadIdx.x >> 5;
  const float* Arow = A + (size_t)mt * 16 * 128;
#pragma unroll
  for (int q = 0; q < 6; ++q) {
    const int jb = (wave * 6 + q) * 16;   // 0..752
    v8f acc = {};
#pragma unroll
    for (int k = 0; k < 4; ++k)
      acc = wmma_bf16(load_a_f32(Arow + k * 32, 128),
                      load_b_bf(Wih + (size_t)jb * 128 + k * 32, 128), acc);
    store_c(gi + (size_t)mt * 16 * 768 + jb, 768, acc);
  }
}

// ---------------------------------------------------------------------------
// Recurrent GRU kernel: one persistent workgroup (16 waves).
// Per step only h @ Whh^T (48 col-tiles, K=256). First 224 weight rows are
// TDM-staged into LDS; the rest stream from L2 (anti-LICM asm stops the
// compiler from hoisting+spilling the loads like round 1).
// ---------------------------------------------------------------------------
#define REC_THREADS 512
#define RZ_LDS_ROWS 224                  // 224*256 bf16 = 112 KB in LDS

__global__ __launch_bounds__(REC_THREADS)
void gru_rec_kernel(const float* __restrict__ gi,   // [2400][768], row n*300+t
                    const float* __restrict__ h0,   // [8][256]
                    const __bf16* __restrict__ Whh, // [768][256]
                    const float* __restrict__ bih,
                    const float* __restrict__ bhh,
                    float* __restrict__ outputs,    // [8][300][256]
                    float* __restrict__ hidden) {   // [8][256]
  __shared__ __bf16 whh_lds[RZ_LDS_ROWS * 256];     // 112 KB
  __shared__ __bf16 hb[16 * 256];                   // 8 KB (rows 8..15 zero)
  __shared__ float  g_h[8 * 768];                   // 24 KB
  __shared__ float  hf[8 * 256];                    // 8 KB

  const int tid  = threadIdx.x;
  const int wave = tid >> 5;

  if (tid == 0)
    tdm_stage_1d(Whh, lds_addr_of(whh_lds), RZ_LDS_ROWS * 256 * 2);

  for (int i = tid; i < 16 * 256; i += REC_THREADS) hb[i] = (__bf16)0.f;
  for (int i = tid; i < 8 * 256; i += REC_THREADS) {
    float h = h0[i];
    hf[i] = h;
    hb[i] = (__bf16)h;
  }
  __builtin_amdgcn_s_wait_tensorcnt(0);
  __syncthreads();

  for (int t = 0; t < 300; ++t) {
    const __bf16* whh_g = Whh;
    asm volatile("" : "+s"(whh_g));      // defeat LICM of weight loads
    // 3 column tiles per wave: jt = wave, wave+16, wave+32 (covers 0..47)
#pragma unroll
    for (int s = 0; s < 3; ++s) {
      const int jb = (wave + s * 16) * 16;
      v8f acc = {};
      if (jb < RZ_LDS_ROWS) {
#pragma unroll
        for (int k = 0; k < 8; ++k)
          acc = wmma_bf16(load_a_bf(hb + k * 32, 256),
                          load_b_bf(whh_lds + (size_t)jb * 256 + k * 32, 256),
                          acc);
      } else {
#pragma unroll
        for (int k = 0; k < 8; ++k)
          acc = wmma_bf16(load_a_bf(hb + k * 32, 256),
                          load_b_bf(whh_g + (size_t)jb * 256 + k * 32, 256),
                          acc);
      }
      store_c_lo(g_h + jb, 768, acc);
    }
    __syncthreads();

    // fused gates (PyTorch order r,z,n); gi streamed from L2
    for (int i = tid; i < 8 * 256; i += REC_THREADS) {
      const int n = i >> 8, e = i & 255;
      const float* gr = gi + (size_t)(n * 300 + t) * 768;
      float r  = sigmoid_f(gr[e]       + bih[e]       + g_h[n*768 + e]       + bhh[e]);
      float z  = sigmoid_f(gr[256 + e] + bih[256 + e] + g_h[n*768 + 256 + e] + bhh[256 + e]);
      float nn = tanh_hw (gr[512 + e] + bih[512 + e] +
                          r * (g_h[n*768 + 512 + e] + bhh[512 + e]));
      float h    = hf[i];
      float hnew = (1.f - z) * nn + z * h;
      hf[i] = hnew;
      hb[i] = (__bf16)hnew;
      outputs[(size_t)(n * 300 + t) * 256 + e] = hnew;
      if (t + 1 < 300) __builtin_prefetch(gr + 768 + e, 0, 0);
    }
    __syncthreads();
  }

  for (int i = tid; i < 8 * 256; i += REC_THREADS) hidden[i] = hf[i];
}

// ---------------------------------------------------------------------------
// Projection C[M][256] = A[M][256] (f32) x Wt^T.  The 128 KB bf16 weight is
// TDM-staged into LDS once per block and reused across PROJ_ROWTILES tiles.
// ---------------------------------------------------------------------------
#define PROJ_ROWTILES 5

__global__ __launch_bounds__(256)
void proj_kernel(const float* __restrict__ A,
                 const __bf16* __restrict__ Wt,   // [256][256]
                 float* __restrict__ C) {
  __shared__ __bf16 wlds[256 * 256];              // 128 KB
  if (threadIdx.x == 0)
    tdm_stage_1d(Wt, lds_addr_of(wlds), 256 * 256 * 2);
  __builtin_amdgcn_s_wait_tensorcnt(0);
  __syncthreads();

  const int wave = threadIdx.x >> 5;
  for (int rt = 0; rt < PROJ_ROWTILES; ++rt) {
    const int mt = blockIdx.x * PROJ_ROWTILES + rt;
    const float* Arow = A + (size_t)mt * 16 * 256;
#pragma unroll
    for (int q = 0; q < 2; ++q) {
      const int jb = (wave * 2 + q) * 16;
      v8f acc = {};
#pragma unroll
      for (int k = 0; k < 8; ++k)
        acc = wmma_bf16(load_a_f32(Arow + k * 32, 256),
                        load_b_bf(wlds + jb * 256 + k * 32, 256), acc);
      store_c(C + (size_t)mt * 16 * 256 + jb, 256, acc);
    }
  }
}

// ---------------------------------------------------------------------------
// Scores + softmax: one block per (n, t) row.  VALU/TRANS bound (98M tanh);
// uses hardware v_tanh_f32.
// ---------------------------------------------------------------------------
__global__ __launch_bounds__(256)
void attn_kernel(const float* __restrict__ w,   // [2400][256]
                 const float* __restrict__ u,   // [8][160][256]
                 const float* __restrict__ v,   // [256]
                 float* __restrict__ attn) {    // [2400][160]
  const int row  = blockIdx.x;
  const int n    = row / 300;
  const int tid  = threadIdx.x;
  const int lane = tid & 31;
  const int wave = tid >> 5;

  __shared__ float wrow[256];
  __shared__ float vv[256];
  __shared__ float sc[160];
  __shared__ float red[2];

  wrow[tid] = w[(size_t)row * 256 + tid];
  vv[tid]   = v[tid];
  __syncthreads();

  const float* ub = u + (size_t)n * 160 * 256;
  for (int x = wave; x < 160; x += 8) {
    const float* up = ub + (size_t)x * 256 + lane * 8;
    if (x + 8 < 160) __builtin_prefetch(ub + (size_t)(x + 8) * 256 + lane * 8, 0, 1);
    float s = 0.f;
#pragma unroll
    for (int i = 0; i < 8; ++i) {
      const int e = lane * 8 + i;
      s += vv[e] * tanh_hw(wrow[e] + up[i]);
    }
#pragma unroll
    for (int off = 16; off; off >>= 1) s += __shfl_xor(s, off, 32);
    if (lane == 0) sc[x] = s;
  }
  __syncthreads();

  if (wave == 0) {
    float m = -3.4e38f;
    for (int x = lane; x < 160; x += 32) m = fmaxf(m, sc[x]);
#pragma unroll
    for (int off = 16; off; off >>= 1) m = fmaxf(m, __shfl_xor(m, off, 32));
    float sum = 0.f;
    for (int x = lane; x < 160; x += 32) sum += __expf(sc[x] - m);
#pragma unroll
    for (int off = 16; off; off >>= 1) sum += __shfl_xor(sum, off, 32);
    if (lane == 0) { red[0] = m; red[1] = sum; }
  }
  __syncthreads();

  const float m = red[0], inv = 1.f / red[1];
  for (int x = tid; x < 160; x += 256)
    attn[(size_t)row * 160 + x] = __expf(sc[x] - m) * inv;
}

// ---------------------------------------------------------------------------
// Host-side orchestration
// ---------------------------------------------------------------------------
extern "C" void kernel_launch(void* const* d_in, const int* in_sizes, int n_in,
                              void* d_out, int out_size, void* d_ws, size_t ws_size,
                              hipStream_t stream) {
  (void)in_sizes; (void)n_in; (void)out_size; (void)ws_size;

  const float* inputs = (const float*)d_in[0];  // [8][300][128]
  const float* memory = (const float*)d_in[1];  // [8][160][256]
  const float* h0     = (const float*)d_in[2];  // [1][8][256]
  const float* Wih    = (const float*)d_in[3];  // [768][128]
  const float* Whh    = (const float*)d_in[4];  // [768][256]
  const float* bih    = (const float*)d_in[5];  // [768]
  const float* bhh    = (const float*)d_in[6];  // [768]
  const float* W      = (const float*)d_in[7];  // [256][256]
  const float* U      = (const float*)d_in[8];  // [256][256]
  const float* v      = (const float*)d_in[9];  // [256]

  float* attn    = (float*)d_out;               // 8*300*160
  float* outputs = attn + 8 * 300 * 160;        // 8*300*256
  float* hidden  = outputs + 8 * 300 * 256;     // 8*256

  char* ws = (char*)d_ws;
  size_t o = 0;
  __bf16* Wih_b = (__bf16*)(ws + o); o += (size_t)768 * 128 * 2;
  __bf16* Whh_b = (__bf16*)(ws + o); o += (size_t)768 * 256 * 2;
  __bf16* W_b   = (__bf16*)(ws + o); o += (size_t)256 * 256 * 2;
  __bf16* U_b   = (__bf16*)(ws + o); o += (size_t)256 * 256 * 2;
  float*  gi_f  = (float*)(ws + o);  o += (size_t)2400 * 768 * 4;
  float*  w_f   = (float*)(ws + o);  o += (size_t)2400 * 256 * 4;
  float*  u_f   = (float*)(ws + o);  o += (size_t)1280 * 256 * 4;

  cvt_kernel<<<(768 * 128 + 255) / 256, 256, 0, stream>>>(Wih, Wih_b, 768 * 128);
  cvt_kernel<<<(768 * 256 + 255) / 256, 256, 0, stream>>>(Whh, Whh_b, 768 * 256);
  cvt_kernel<<<(256 * 256 + 255) / 256, 256, 0, stream>>>(W, W_b, 256 * 256);
  cvt_kernel<<<(256 * 256 + 255) / 256, 256, 0, stream>>>(U, U_b, 256 * 256);

  gi_kernel<<<150, 256, 0, stream>>>(inputs, Wih_b, gi_f);

  gru_rec_kernel<<<1, REC_THREADS, 0, stream>>>(gi_f, h0, Whh_b,
                                                bih, bhh, outputs, hidden);

  proj_kernel<<<30, 256, 0, stream>>>(outputs, W_b, w_f);  // 150 row tiles
  proj_kernel<<<16, 256, 0, stream>>>(memory,  U_b, u_f);  // 80 row tiles

  attn_kernel<<<2400, 256, 0, stream>>>(w_f, u_f, v, attn);
}
